// SSM_gray_con_56994216018400
// MI455X (gfx1250) — compile-verified, matrix operand order checked by reading, not run on hardware
//
#include <hip/hip_runtime.h>
#include <math.h>

// ---------------- CDNA5 vector types ----------------
typedef __attribute__((ext_vector_type(16))) _Float16 v16h;
typedef __attribute__((ext_vector_type(8)))  float    v8f;

// ---------------- problem dims ----------------
#define NXd 128
#define NYd 64
#define NMd 32
#define NUd 32
#define NSTEPS 1024
#define Bd 128
#define TPd 64

// ---------------- d_out offsets (floats) ----------------
static constexpr size_t OFF_X     = 0;
static constexpr size_t OFF_Y     = 16777216;
static constexpr size_t OFF_U     = 25165824;
static constexpr size_t OFF_SXMIN = 29360128;
static constexpr size_t OFF_SXMAX = 46137344;
static constexpr size_t OFF_SUMIN = 62914560;
static constexpr size_t OFF_SUMAX = 67108864;
static constexpr size_t OFF_SDXX  = 71303168;  // also used to stage bias = Bu+Ed
static constexpr size_t OFF_SDXU  = 88080384;
static constexpr size_t OFF_SDXD  = 104857600;
static constexpr size_t OFF_SERR  = 121634816;

// ---------------- ws layout (halves, then floats) ----------------
#define WS_WA   0        // 8 tiles x 4 chunks  (W_A   128x128)
#define WS_WC   16384    // 4 x 4               (W_C   128x64)
#define WS_WE   24576    // 8 x 1               (W_E    32x128)
#define WS_WB   28672    // 8 x 1               (W_Bpos 32x128)
#define WS_WIH  32768    // 8 x 2               (W_ih   64x128)
#define WS_WHH  40960    // 8 x 4               (W_hh  128x128)
#define WS_WHF  57344    // 2 x 32              (W_hf' 1024x32)
#define WS_HALVES 90112

// ---------------- helpers ----------------
__device__ __forceinline__ float sigm(float x) { return 1.0f / (1.0f + __expf(-x)); }
__device__ __forceinline__ float relu_(float x) { return fmaxf(x, 0.0f); }
// branch-free tanh-approx gelu (jax default approximate=True)
__device__ __forceinline__ float gelu_tanh(float x) {
  float z = 0.7978845608028654f * (x + 0.044715f * x * x * x);
  float e = __expf(2.0f * z);
  float th = (e - 1.0f) / (e + 1.0f);
  return 0.5f * x * (1.0f + th);
}
// K index held by (pair r, lane-half) per CDNA5 16-bit A-matrix 16x32 layout
__device__ __forceinline__ int patk(int r, int half) {
  return (r < 4) ? (8 * half + 2 * r) : (16 + 8 * half + 2 * (r - 4));
}
__device__ __forceinline__ v8f wmma16(v16h a, v16h b, v8f c) {
  return __builtin_amdgcn_wmma_f32_16x16x32_f16(false, a, false, b, (short)0, c,
                                                false, false);
}
// Build A operand (16x32 fp16 tile) from row-major fp16 LDS
// (the patk pattern makes this two contiguous 16B runs -> ds_load_b128 x2)
__device__ __forceinline__ v16h ldsA_h(const _Float16* base, int ldk, int kbase,
                                       int row, int half) {
  v16h a;
#pragma unroll
  for (int r = 0; r < 8; ++r) {
    int pk = patk(r, half);
    a[2 * r]     = base[row * ldk + kbase + pk];
    a[2 * r + 1] = base[row * ldk + kbase + pk + 1];
  }
  return a;
}
// Same but sourcing fp32 LDS (converts on the fly)
__device__ __forceinline__ v16h ldsA_f(const float* base, int ldk, int kbase,
                                       int row, int half) {
  v16h a;
#pragma unroll
  for (int r = 0; r < 8; ++r) {
    int pk = patk(r, half);
    a[2 * r]     = (_Float16)base[row * ldk + kbase + pk];
    a[2 * r + 1] = (_Float16)base[row * ldk + kbase + pk + 1];
  }
  return a;
}
// Scatter one weight element into the per-lane-contiguous B-operand blob
__device__ __forceinline__ void blob_write(_Float16* blob, int nchunks, int k,
                                           int n, float v) {
  int tile = n >> 4, nn = n & 15;
  int chunk = k >> 5;
  int kr = k & 31;
  int k2 = kr >> 1, bb = kr & 1;
  int half = (k2 >> 2) & 1;
  int r = (k2 & 3) + 4 * (k2 >> 3);
  int lane = nn + 16 * half;
  blob[(((tile * nchunks + chunk) * 32) + lane) * 16 + 2 * r + bb] = (_Float16)v;
}

// ---------------- CDNA5 async global->LDS copy (ASYNCcnt path) ----------------
// VGLOBAL encoding: VDST = LDS byte offset VGPR, VADDR = 64-bit global address.
__device__ __forceinline__ void async_ld_b128(unsigned lds_off, const void* g) {
  asm volatile("global_load_async_to_lds_b128 %0, %1, off"
               :: "v"(lds_off), "v"(g)
               : "memory");
}
__device__ __forceinline__ void wait_async0() {
  asm volatile("s_wait_asynccnt 0x0" ::: "memory");
}
// generic (flat) LDS pointer -> 32-bit LDS byte offset
__device__ __forceinline__ unsigned lds_off32(const void* p) {
  return (unsigned)(unsigned long long)p;
}

// ---------------- prep: zero error accumulator ----------------
__global__ void k_init_err(float* errws) {
  if (threadIdx.x == 0 && blockIdx.x == 0) errws[0] = 0.0f;
}

// ---------------- prep: spectral orthogonality error ----------------
__global__ void k_err(const float* __restrict__ U_A, const float* __restrict__ V_A,
                      const float* __restrict__ U_C, const float* __restrict__ V_C,
                      float* __restrict__ errws) {
  __shared__ float red[256];
  int gid = blockIdx.x * 256 + threadIdx.x;  // exactly 53248 entries
  float term = 0.0f;
  if (gid < 16384) {
    int i = gid >> 7, j = gid & 127;
    float d = 0.0f;
    for (int k = 0; k < 128; ++k) d += U_A[i * 128 + k] * U_A[j * 128 + k];
    d -= (i == j) ? 1.0f : 0.0f;
    term = d * d * (1.0f / 16384.0f);
  } else if (gid < 32768) {
    int e = gid - 16384; int i = e >> 7, j = e & 127;
    float d = 0.0f;
    for (int k = 0; k < 128; ++k) d += V_A[i * 128 + k] * V_A[j * 128 + k];
    d -= (i == j) ? 1.0f : 0.0f;
    term = d * d * (1.0f / 16384.0f);
  } else if (gid < 49152) {
    int e = gid - 32768; int i = e >> 7, j = e & 127;
    float d = 0.0f;
    for (int k = 0; k < 128; ++k) d += U_C[i * 128 + k] * U_C[j * 128 + k];
    d -= (i == j) ? 1.0f : 0.0f;
    term = d * d * (1.0f / 16384.0f);
  } else {
    int e = gid - 49152; int i = e >> 6, j = e & 63;
    float d = 0.0f;
    for (int k = 0; k < 64; ++k) d += V_C[i * 64 + k] * V_C[j * 64 + k];
    d -= (i == j) ? 1.0f : 0.0f;
    term = d * d * (1.0f / 4096.0f);
  }
  red[threadIdx.x] = term;
  __syncthreads();
  for (int s = 128; s > 0; s >>= 1) {
    if (threadIdx.x < s) red[threadIdx.x] += red[threadIdx.x + s];
    __syncthreads();
  }
  if (threadIdx.x == 0) atomicAdd(errws, red[0]);
}

__global__ void k_bcast(const float* __restrict__ errws, float* __restrict__ out) {
  int gid = blockIdx.x * blockDim.x + threadIdx.x;
  if (gid < NSTEPS) out[OFF_SERR + gid] = errws[0];
}

// ---------------- prep: build all fp16 B-operand blobs ----------------
__global__ void k_pack(const float* __restrict__ U_A, const float* __restrict__ sig_A,
                       const float* __restrict__ V_A, const float* __restrict__ U_C,
                       const float* __restrict__ sig_C, const float* __restrict__ V_C,
                       const float* __restrict__ w_E, const float* __restrict__ s_E,
                       const float* __restrict__ w_B, const float* __restrict__ W_hf,
                       const float* __restrict__ W_ih, const float* __restrict__ W_hh,
                       _Float16* __restrict__ wsh) {
  int gid = blockIdx.x * 256 + threadIdx.x;  // 90112 tasks
  if (gid < 16384) {                         // W_A = (U_A * sA) @ V_A
    int ki = gid >> 7, n = gid & 127;
    float acc = 0.0f;
    for (int c = 0; c < 128; ++c) {
      float s = 0.1f + 0.8f * sigm(sig_A[c]);
      acc += U_A[ki * 128 + c] * s * V_A[c * 128 + n];
    }
    blob_write(wsh + WS_WA, 4, ki, n, acc);
  } else if (gid < 24576) {                  // W_C = (U_C[:, :64] * sC) @ V_C
    int e = gid - 16384; int ki = e >> 6, n = e & 63;
    float acc = 0.0f;
    for (int c = 0; c < 64; ++c) {
      float s = 0.9f + 0.1f * sigm(sig_C[c]);
      acc += U_C[ki * 128 + c] * s * V_C[c * 64 + n];
    }
    blob_write(wsh + WS_WC, 4, ki, n, acc);
  } else if (gid < 28672) {                  // W_E = s * softmax_axis0(w_E)
    int e = gid - 24576; int ki = e >> 7, n = e & 127;
    float mx = -1e30f;
    for (int r = 0; r < 32; ++r) mx = fmaxf(mx, w_E[r * 128 + n]);
    float sum = 0.0f;
    for (int r = 0; r < 32; ++r) sum += __expf(w_E[r * 128 + n] - mx);
    float s = 1.0f - 0.95f * sigm(s_E[ki * 128 + n]);
    blob_write(wsh + WS_WE, 1, ki, n, s * __expf(w_E[ki * 128 + n] - mx) / sum);
  } else if (gid < 32768) {                  // W_Bpos = relu(w_B)
    int e = gid - 28672; int ki = e >> 7, n = e & 127;
    blob_write(wsh + WS_WB, 1, ki, n, relu_(w_B[ki * 128 + n]));
  } else if (gid < 40960) {                  // W_ih
    int e = gid - 32768; int ki = e >> 7, n = e & 127;
    blob_write(wsh + WS_WIH, 2, ki, n, W_ih[ki * 128 + n]);
  } else if (gid < 57344) {                  // W_hh
    int e = gid - 40960; int ki = e >> 7, n = e & 127;
    blob_write(wsh + WS_WHH, 4, ki, n, W_hh[ki * 128 + n]);
  } else if (gid < 90112) {                  // W_hf' : [i*32+j][o] = W_hf[o][i][j]
    int e = gid - 57344; int kbig = e >> 5, o = e & 31;
    int i = kbig >> 5, j = kbig & 31;
    blob_write(wsh + WS_WHF, 32, kbig, o, W_hf[(o * 32 + i) * 32 + j]);
  }
}

// ---------------- RNN state estimator (64 serial steps, 8 WGs over batch) ----------------
__global__ void k_rnn(const float* __restrict__ Ym, const _Float16* __restrict__ wih,
                      const _Float16* __restrict__ whh, float* __restrict__ x0ws) {
  __shared__ _Float16 h_h[16 * 128];
  __shared__ _Float16 ym_h[16 * 64];
  const int tid = threadIdx.x;
  const int wave = tid >> 5, lane = tid & 31;
  const int la = lane & 15, hf = lane >> 4;
  const int b0 = blockIdx.x * 16;
  const v16h* wihv = (const v16h*)wih;
  const v16h* whhv = (const v16h*)whh;
  v16h bih[2], bhh[4];
#pragma unroll
  for (int c = 0; c < 2; ++c) bih[c] = wihv[(wave * 2 + c) * 32 + lane];
#pragma unroll
  for (int c = 0; c < 4; ++c) bhh[c] = whhv[(wave * 4 + c) * 32 + lane];
  for (int i = tid; i < 16 * 128; i += 256) h_h[i] = (_Float16)0.0f;
  __syncthreads();
  v8f g = {};
  for (int t = 0; t < TPd; ++t) {
    for (int i = tid; i < 16 * 64; i += 256) {
      int row = i >> 6, col = i & 63;
      ym_h[i] = (_Float16)Ym[((size_t)t * Bd + b0 + row) * NYd + col];
    }
    __syncthreads();
    v8f c = {};
#pragma unroll
    for (int c0 = 0; c0 < 2; ++c0)
      c = wmma16(ldsA_h(ym_h, 64, 32 * c0, la, hf), bih[c0], c);
#pragma unroll
    for (int c1 = 0; c1 < 4; ++c1)
      c = wmma16(ldsA_h(h_h, 128, 32 * c1, la, hf), bhh[c1], c);
#pragma unroll
    for (int r = 0; r < 8; ++r) g[r] = gelu_tanh(c[r]);
    __syncthreads();
#pragma unroll
    for (int r = 0; r < 8; ++r)
      h_h[(r + 8 * hf) * 128 + wave * 16 + la] = (_Float16)g[r];
    __syncthreads();
  }
#pragma unroll
  for (int r = 0; r < 8; ++r)
    x0ws[(size_t)(b0 + r + 8 * hf) * NXd + wave * 16 + la] = g[r];
}

// ---------------- parallel per-step part: u, Bu, Ed, slacks, bias staging ----------------
__global__ void k_u(const float* __restrict__ Mf, const float* __restrict__ DT,
                    const float* __restrict__ Dd, const float* __restrict__ UMIN,
                    const float* __restrict__ UMAX, const float* __restrict__ dxmin,
                    const float* __restrict__ dxmax, const _Float16* __restrict__ whf,
                    const _Float16* __restrict__ wB, const _Float16* __restrict__ wE,
                    float* __restrict__ out) {
  // stage = [ m (16x32) | dT (16x32) | d (16x32) ] fp32
  __shared__ __align__(16) float stage[3 * 512];
  __shared__ _Float16 u_h[16 * 32];
  __shared__ float cpart[8 * 32 * 8];
  const int tid = threadIdx.x;
  const int wave = tid >> 5, lane = tid & 31;
  const int la = lane & 15, hf = lane >> 4;
  const int t = blockIdx.x;
  const int b0 = blockIdx.y * 16;
  const float* m_s  = stage;
  const float* dT_s = stage + 512;
  const float* d_s  = stage + 1024;

  // --- async global->LDS staging (16B per lane, wave-uniform branches) ---
  {
    const float* mbase = Mf + ((size_t)t * Bd + b0) * 32;
    const float* tbase = DT + ((size_t)t * Bd + b0) * 32;
    const float* dbase = Dd + ((size_t)t * Bd + b0) * 32;
    unsigned lbase = lds_off32(stage);
    if (tid < 128) {                       // waves 0..3 : m
      async_ld_b128(lbase + (unsigned)tid * 16u, mbase + tid * 4);
    } else {                               // waves 4..7 : dT
      unsigned i = (unsigned)(tid - 128);
      async_ld_b128(lbase + 2048u + i * 16u, tbase + i * 4);
    }
    if (tid < 128) {                       // waves 0..3 : d
      async_ld_b128(lbase + 4096u + (unsigned)tid * 16u, dbase + tid * 4);
    }
    wait_async0();
  }
  __syncthreads();

  // u = outer(m, dT) @ W_hf'  as 16x32x1024 GEMM; K split over 4 wave pairs
  {
    const v16h* whfv = (const v16h*)whf;
    int tile = wave & 1;
    int kk0 = (wave >> 1) * 8;
    v8f c = {};
#pragma unroll
    for (int q = 0; q < 8; ++q) {
      int kk = kk0 + q;
      float mval = m_s[la * 32 + kk];  // A[m, 32*kk + j] = m[m,kk] * dT[m,j]
      v16h a;
#pragma unroll
      for (int r = 0; r < 8; ++r) {
        int pk = patk(r, hf);
        a[2 * r]     = (_Float16)(mval * dT_s[la * 32 + pk]);
        a[2 * r + 1] = (_Float16)(mval * dT_s[la * 32 + pk + 1]);
      }
      v16h b = whfv[(tile * 32 + kk) * 32 + lane];
      c = wmma16(a, b, c);
    }
#pragma unroll
    for (int r = 0; r < 8; ++r) cpart[wave * 256 + lane * 8 + r] = c[r];
  }
  __syncthreads();

  if (wave < 2) {  // reduce K partials; emit U / Sumin / Sumax
    int tile = wave;
    v8f u = {};
#pragma unroll
    for (int w2 = 0; w2 < 4; ++w2)
#pragma unroll
      for (int r = 0; r < 8; ++r) u[r] += cpart[(tile + 2 * w2) * 256 + lane * 8 + r];
    int col = tile * 16 + la;
#pragma unroll
    for (int r = 0; r < 8; ++r) {
      int row = r + 8 * hf;
      size_t gi = ((size_t)t * Bd + b0 + row) * 32 + col;
      float uu = u[r];
      out[OFF_U + gi] = uu;
      out[OFF_SUMIN + gi] = relu_(UMIN[gi] - uu);
      out[OFF_SUMAX + gi] = relu_(uu - UMAX[gi]);
      u_h[row * 32 + col] = (_Float16)uu;
    }
  }
  __syncthreads();

  // Bu = u @ W_Bpos, Ed = d @ W_E ; stage bias into Sdx_x region
  {
    const v16h* wBv = (const v16h*)wB;
    const v16h* wEv = (const v16h*)wE;
    v16h bB = wBv[wave * 32 + lane];
    v16h bE = wEv[wave * 32 + lane];
    v8f cb = {}, ce = {};
    cb = wmma16(ldsA_h(u_h, 32, 0, la, hf), bB, cb);
    ce = wmma16(ldsA_f(d_s, 32, 0, la, hf), bE, ce);
    int n = wave * 16 + la;
    float dmn = dxmin[n], dmx = dxmax[n];
#pragma unroll
    for (int r = 0; r < 8; ++r) {
      int row = r + 8 * hf;
      size_t gi = ((size_t)t * Bd + b0 + row) * NXd + n;
      float buv = cb[r], edv = ce[r];
      out[OFF_SDXX + gi] = buv + edv;  // bias; overwritten by k_scan with dx_x
      out[OFF_SDXU + gi] = relu_(dmn - buv) + relu_(buv - dmx);
      out[OFF_SDXD + gi] = relu_(dmn - edv) + relu_(edv - dmx);
    }
  }
}

// ---------------- serial scan: x_{t+1} = xc@W_A + bias ; y = x@W_C ----------------
__global__ void k_scan(const float* __restrict__ XMIN, const float* __restrict__ XMAX,
                       const float* __restrict__ x0c, const _Float16* __restrict__ wA,
                       const _Float16* __restrict__ wC, const float* __restrict__ x0ws,
                       float* __restrict__ out) {
  __shared__ _Float16 xc_h[16 * 128];
  __shared__ _Float16 xn_h[16 * 128];
  const int tid = threadIdx.x;
  const int wave = tid >> 5, lane = tid & 31;
  const int la = lane & 15, hf = lane >> 4;
  const int b0 = blockIdx.x * 16;
  const int n0 = wave * 16;
  const v16h* wAv = (const v16h*)wA;
  const v16h* wCv = (const v16h*)wC;
  v16h bA[4];
#pragma unroll
  for (int c = 0; c < 4; ++c) bA[c] = wAv[(wave * 4 + c) * 32 + lane];
  v16h bC[4];
#pragma unroll
  for (int c = 0; c < 4; ++c)
#pragma unroll
    for (int e = 0; e < 16; ++e) bC[c][e] = (_Float16)0.0f;
  if (wave < 4) {
#pragma unroll
    for (int c = 0; c < 4; ++c) bC[c] = wCv[(wave * 4 + c) * 32 + lane];
  }
  const float xcl = x0c[n0 + la];
  float xprev[8];
#pragma unroll
  for (int r = 0; r < 8; ++r) {
    int row = r + 8 * hf;
    float v = x0ws[(size_t)(b0 + row) * NXd + n0 + la];
    xprev[r] = v;
    xc_h[row * 128 + n0 + la] = (_Float16)(v + xcl);
  }
  __syncthreads();

  for (int t = 0; t < NSTEPS; ++t) {
    v8f c;
#pragma unroll
    for (int r = 0; r < 8; ++r) {  // accumulator seeded with bias = Bu+Ed
      int row = r + 8 * hf;
      c[r] = out[OFF_SDXX + ((size_t)t * Bd + b0 + row) * NXd + n0 + la];
    }
#pragma unroll
    for (int ck = 0; ck < 4; ++ck)
      c = wmma16(ldsA_h(xc_h, 128, 32 * ck, la, hf), bA[ck], c);
    if (t + 1 < NSTEPS) {  // pull next step's lines while WMMAs drain
      __builtin_prefetch(&out[OFF_SDXX + ((size_t)(t + 1) * Bd + b0 + (tid >> 4)) * NXd], 0, 1);
      __builtin_prefetch(&XMIN[((size_t)(t + 1) * Bd + b0 + (tid >> 4)) * NXd], 0, 1);
    }
#pragma unroll
    for (int r = 0; r < 8; ++r) {
      int row = r + 8 * hf;
      size_t gi = ((size_t)t * Bd + b0 + row) * NXd + n0 + la;
      float xn = c[r];
      out[OFF_X + gi] = xn;
      out[OFF_SXMIN + gi] = relu_(XMIN[gi] - xn);
      out[OFF_SXMAX + gi] = relu_(xn - XMAX[gi]);
      out[OFF_SDXX + gi] = xn - xprev[r];
      xprev[r] = xn;
    }
    __syncthreads();  // all waves done reading xc_h/xn_h
#pragma unroll
    for (int r = 0; r < 8; ++r) {
      int row = r + 8 * hf;
      xn_h[row * 128 + n0 + la] = (_Float16)xprev[r];
      xc_h[row * 128 + n0 + la] = (_Float16)(xprev[r] + xcl);
    }
    __syncthreads();
    if (wave < 4) {  // y = x_new @ W_C  (N=64 -> waves 0..3)
      v8f cy = {};
#pragma unroll
      for (int ck = 0; ck < 4; ++ck)
        cy = wmma16(ldsA_h(xn_h, 128, 32 * ck, la, hf), bC[ck], cy);
#pragma unroll
      for (int r = 0; r < 8; ++r) {
        int row = r + 8 * hf;
        out[OFF_Y + ((size_t)t * Bd + b0 + row) * NYd + n0 + la] = cy[r];
      }
    }
    __syncthreads();
  }
}

// ---------------- launch ----------------
extern "C" void kernel_launch(void* const* d_in, const int* in_sizes, int n_in,
                              void* d_out, int out_size, void* d_ws, size_t ws_size,
                              hipStream_t stream) {
  (void)in_sizes; (void)n_in; (void)out_size; (void)ws_size;
  const float* Ym   = (const float*)d_in[0];
  const float* Mf   = (const float*)d_in[1];
  const float* DT   = (const float*)d_in[2];
  const float* Dd   = (const float*)d_in[3];
  const float* XMIN = (const float*)d_in[4];
  const float* XMAX = (const float*)d_in[5];
  const float* UMIN = (const float*)d_in[6];
  const float* UMAX = (const float*)d_in[7];
  const float* x0c  = (const float*)d_in[8];
  const float* U_A  = (const float*)d_in[9];
  const float* sigA = (const float*)d_in[10];
  const float* V_A  = (const float*)d_in[11];
  const float* U_C  = (const float*)d_in[12];
  const float* sigC = (const float*)d_in[13];
  const float* V_C  = (const float*)d_in[14];
  const float* w_E  = (const float*)d_in[15];
  const float* s_E  = (const float*)d_in[16];
  const float* w_B  = (const float*)d_in[17];
  const float* W_hf = (const float*)d_in[18];
  const float* W_ih = (const float*)d_in[19];
  const float* W_hh = (const float*)d_in[20];
  const float* dxmn = (const float*)d_in[21];
  const float* dxmx = (const float*)d_in[22];
  float* out = (float*)d_out;

  _Float16* wsh = (_Float16*)d_ws;
  float* fbase = (float*)(wsh + WS_HALVES);
  float* x0ws  = fbase;          // 128*128 floats
  float* errws = fbase + 16384;  // 1 float

  k_init_err<<<1, 64, 0, stream>>>(errws);
  k_err<<<208, 256, 0, stream>>>(U_A, V_A, U_C, V_C, errws);
  k_pack<<<352, 256, 0, stream>>>(U_A, sigA, V_A, U_C, sigC, V_C, w_E, s_E, w_B,
                                  W_hf, W_ih, W_hh, wsh);
  k_bcast<<<4, 256, 0, stream>>>(errws, out);
  k_rnn<<<8, 256, 0, stream>>>(Ym, wsh + WS_WIH, wsh + WS_WHH, x0ws);
  k_u<<<dim3(NSTEPS, 8), 256, 0, stream>>>(Mf, DT, Dd, UMIN, UMAX, dxmn, dxmx,
                                           wsh + WS_WHF, wsh + WS_WB, wsh + WS_WE,
                                           out);
  k_scan<<<8, 256, 0, stream>>>(XMIN, XMAX, x0c, wsh + WS_WA, wsh + WS_WC, x0ws, out);
}
